// GroupedQueryAttention_5514738008348
// MI455X (gfx1250) — compile-verified, matrix operand order checked by reading, not run on hardware
//
#include <hip/hip_runtime.h>
#include <hip/hip_bf16.h>
#include <stdint.h>

// ---------------------------------------------------------------------------
// Problem constants (reference: B=2, S=2048, D=1024, H=16, KV=8, HD=64, W=512)
// ---------------------------------------------------------------------------
#define B_   2
#define S_   2048
#define D_   1024
#define H_   16
#define KV_  8
#define HD_  64
#define WIN_ 512

typedef __attribute__((ext_vector_type(16))) __bf16 v16bf;
typedef __attribute__((ext_vector_type(8)))  float  v8f;

union Frag16 {
  unsigned int   w[8];
  unsigned short u[16];
  v16bf          v;
};

__device__ __forceinline__ unsigned short f2bf(float f) {
  unsigned int u = __float_as_uint(f);
  u = (u + 0x7fffu + ((u >> 16) & 1u)) >> 16;   // round-to-nearest-even
  return (unsigned short)u;
}
__device__ __forceinline__ unsigned int ld_u32(const void* p) {
  unsigned int w; __builtin_memcpy(&w, p, 4); return w;
}
// K-offset of element pair j in the 16-bit A/B fragment layout
// (ISA 7.12.2: V0..V3 hold K=0..7 (+8 for hi half-lanes), V4..V7 hold K=16..23 (+8))
__device__ __forceinline__ int kbase(int j, int hi) {
  return ((j & 3) << 1) + ((j >= 4) ? 16 : 0) + (hi ? 8 : 0);
}

// CDNA5 async copy: 16B global -> LDS, tracked by ASYNCcnt (ISA 10.x / 15.18.3).
// lds_off = raw LDS byte offset (flat aperture maps LDS offset to addr[31:0]).
__device__ __forceinline__ void async_ld_b128(unsigned lds_off, const void* gptr) {
  asm volatile("global_load_async_to_lds_b128 %0, %1, off"
               :: "v"(lds_off), "v"(gptr) : "memory");
}
__device__ __forceinline__ void wait_asynccnt0() {
  asm volatile("s_wait_asynccnt 0" ::: "memory");
}

// ---------------------------------------------------------------------------
// fp32 -> bf16 conversion
// ---------------------------------------------------------------------------
__global__ void k_f32_to_bf16(const float* __restrict__ src,
                              unsigned short* __restrict__ dst, int n) {
  int i = blockIdx.x * 256 + threadIdx.x;
  if (i < n) dst[i] = f2bf(src[i]);
}

// ---------------------------------------------------------------------------
// bf16 GEMM: C[M,N](f32) = A[M,K](bf16,row-major) @ B[K,N](bf16,row-major)
// block = 256 threads (8 waves), block tile 128x64, k-step 32.
// A tile staged with global_load_async_to_lds_b128 (ASYNCcnt path);
// B tile staged transposed through VGPRs (needs the transpose anyway).
// ---------------------------------------------------------------------------
__global__ __launch_bounds__(256) void k_gemm_bf16(
    const unsigned short* __restrict__ A, const unsigned short* __restrict__ Bm,
    float* __restrict__ C, int M, int N, int K) {
  __shared__ __align__(16) unsigned short As[128 * 32];   // A tile, row-major
  __shared__ __align__(16) unsigned short Bt[64 * 32];    // B tile, transposed [n][k]

  const int tid  = threadIdx.x;
  const int wave = tid >> 5, lane = tid & 31;
  const int hi   = (lane >= 16) ? 1 : 0;
  const int mcol = lane & 15;
  const int m0 = blockIdx.y * 128, n0 = blockIdx.x * 64;

  v8f acc[4];
  for (int t = 0; t < 4; t++)
    for (int r = 0; r < 8; r++) acc[t][r] = 0.0f;

  const int arow = tid >> 1, acb = (tid & 1) * 16;   // A loader: half-row each
  const int bn = tid & 63, bkb = (tid >> 6) * 8;     // B loader: 8 k's per thread

  const unsigned alds0 = (unsigned)(size_t)&As[arow * 32 + acb];
  const unsigned alds1 = (unsigned)(size_t)&As[arow * 32 + acb + 8];

  for (int kc = 0; kc < K; kc += 32) {
    __syncthreads();
    {  // stage A tile 128x32 via async DMA to LDS (two 16B transfers/thread)
      const unsigned short* gp = A + (size_t)(m0 + arow) * K + kc + acb;
      async_ld_b128(alds0, gp);
      async_ld_b128(alds1, gp + 8);
    }
    // stage B tile 32x64, transposed into LDS (coalesced global reads along n)
    for (int i = 0; i < 8; i++)
      Bt[bn * 32 + bkb + i] = Bm[(size_t)(kc + bkb + i) * N + n0 + bn];
    wait_asynccnt0();
    __syncthreads();

    Frag16 fa;
    for (int j = 0; j < 8; j++)
      fa.w[j] = ld_u32(&As[(wave * 16 + mcol) * 32 + kbase(j, hi)]);
    for (int t = 0; t < 4; t++) {
      Frag16 fb;
      for (int j = 0; j < 8; j++)
        fb.w[j] = ld_u32(&Bt[(t * 16 + mcol) * 32 + kbase(j, hi)]);
      acc[t] = __builtin_amdgcn_wmma_f32_16x16x32_bf16(
          false, fa.v, false, fb.v, (short)0, acc[t], false, false);
    }
  }

  for (int t = 0; t < 4; t++)
    for (int r = 0; r < 8; r++) {
      int row  = m0 + wave * 16 + r + hi * 8;
      int colc = n0 + t * 16 + mcol;
      C[(size_t)row * N + colc] = acc[t][r];
    }
}

// ---------------------------------------------------------------------------
// RoPE on Q (fp32 in, bf16 head-major out [b][h][s][hd])
// ---------------------------------------------------------------------------
__global__ void k_rope_q(const float* __restrict__ Y, unsigned short* __restrict__ Qb) {
  int idx = blockIdx.x * 256 + threadIdx.x;     // B*S*H*HD = 2^22 threads
  int d = idx & 63;
  int h = (idx >> 6) & 15;
  int s = (idx >> 10) & 2047;
  int b = idx >> 21;
  const float* row = Y + (size_t)(b * S_ + s) * (H_ * HD_) + h * HD_;
  float x  = row[d];
  float xp = row[d ^ 32];
  float inv = __powf(10000.0f, -(float)(d & 31) * (1.0f / 32.0f));
  float ang = (float)s * inv;
  float sn, cs; __sincosf(ang, &sn, &cs);
  float rot = (d < 32) ? -xp : xp;
  Qb[((size_t)(b * H_ + h) * S_ + s) * HD_ + d] = f2bf(x * cs + rot * sn);
}

// RoPE on K (head-major [b][kv][s][hd]) + convert V TRANSPOSED ([b][kv][hd][s])
// so the P@V B-fragments become contiguous dword/b128 loads.
__global__ void k_rope_kv(const float* __restrict__ Yk, const float* __restrict__ Yv,
                          unsigned short* __restrict__ Kb, unsigned short* __restrict__ Vt) {
  int idx = blockIdx.x * 256 + threadIdx.x;     // B*S*KV*HD = 2^21 threads
  int d  = idx & 63;
  int kv = (idx >> 6) & 7;
  int s  = (idx >> 9) & 2047;
  int b  = idx >> 20;
  size_t src = (size_t)(b * S_ + s) * (KV_ * HD_) + kv * HD_;
  float x  = Yk[src + d];
  float xp = Yk[src + (d ^ 32)];
  float inv = __powf(10000.0f, -(float)(d & 31) * (1.0f / 32.0f));
  float ang = (float)s * inv;
  float sn, cs; __sincosf(ang, &sn, &cs);
  float rot = (d < 32) ? -xp : xp;
  Kb[((size_t)(b * KV_ + kv) * S_ + s) * HD_ + d] = f2bf(x * cs + rot * sn);
  Vt[((size_t)(b * KV_ + kv) * HD_ + d) * S_ + s] = f2bf(Yv[src + d]);
}

// ---------------------------------------------------------------------------
// Flash attention with sliding window. One wave per 16-query tile, 4 waves per
// block. Scores: Q(16xHD) x K^T via 2x v_wmma per 16-key tile. P is transposed
// to A-layout through a per-wave LDS tile, then P @ V via 4x v_wmma with
// b128 fragment loads from the transposed V.
// ---------------------------------------------------------------------------
__global__ __launch_bounds__(128) void k_attn(
    const unsigned short* __restrict__ Qb, const unsigned short* __restrict__ Kb,
    const unsigned short* __restrict__ Vt, unsigned short* __restrict__ Ob) {
  __shared__ __align__(16) unsigned short Plds[4][16 * 32];

  const int tid  = threadIdx.x;
  const int wave = tid >> 5, lane = tid & 31;
  const int hi   = (lane >= 16) ? 1 : 0;
  const int col  = lane & 15;

  int tileIdx = blockIdx.x * 4 + wave;          // B*H*(S/16) = 4096 tiles
  int qt = tileIdx & 127;
  int h  = (tileIdx >> 7) & 15;
  int b  = tileIdx >> 11;
  int q0 = qt * 16;
  int kvh = h >> 1;                             // H/KV = 2 queries per kv head

  const unsigned short* Qhead = Qb + ((size_t)(b * H_ + h) * S_) * HD_;
  const unsigned short* Khead = Kb + ((size_t)(b * KV_ + kvh) * S_) * HD_;
  const unsigned short* Vhead = Vt + ((size_t)(b * KV_ + kvh) * HD_) * S_;

  // Q A-fragments for d-chunks [0,32) and [32,64)
  Frag16 qa[2];
  {
    const unsigned short* qrow = Qhead + (size_t)(q0 + col) * HD_;
    for (int half = 0; half < 2; half++)
      for (int j = 0; j < 8; j++)
        qa[half].w[j] = ld_u32(qrow + half * 32 + kbase(j, hi));
  }

  float mrow[8], lrow[8];
  v8f o[4];
  for (int r = 0; r < 8; r++) { mrow[r] = -3.0e38f; lrow[r] = 0.0f; }
  for (int t = 0; t < 4; t++)
    for (int r = 0; r < 8; r++) o[t][r] = 0.0f;

  int kmin = q0 - WIN_; if (kmin < 0) kmin = 0;
  int kc0 = kmin & ~31;
  for (int kc = kc0; kc < q0 + 16; kc += 32) {
    // ---- scores: two 16x16 tiles (keys kc..kc+15 and kc+16..kc+31) ----
    v8f s0, s1;
    for (int r = 0; r < 8; r++) { s0[r] = 0.0f; s1[r] = 0.0f; }
    for (int kt = 0; kt < 2; kt++) {
      int key = kc + kt * 16 + col;
      if (key > S_ - 1) key = S_ - 1;           // clamped rows get masked below
      const unsigned short* krow = Khead + (size_t)key * HD_;
      v8f* dst = kt ? &s1 : &s0;
      for (int half = 0; half < 2; half++) {
        Frag16 fb;
        for (int j = 0; j < 8; j++)
          fb.w[j] = ld_u32(krow + half * 32 + kbase(j, hi));
        *dst = __builtin_amdgcn_wmma_f32_16x16x32_bf16(
            false, qa[half].v, false, fb.v, (short)0, *dst, false, false);
      }
    }

    // ---- scale, mask, online softmax, stage P into LDS ----
    for (int r = 0; r < 8; r++) {
      int qi  = q0 + r + hi * 8;
      int kj0 = kc + col;
      int kj1 = kj0 + 16;
      float x0 = s0[r] * 0.125f;                // 1/sqrt(64)
      float x1 = s1[r] * 0.125f;
      if (kj0 > qi || kj0 < qi - WIN_) x0 = -3.0e38f;
      if (kj1 > qi || kj1 < qi - WIN_) x1 = -3.0e38f;
      float mx = fmaxf(x0, x1);
      for (int m2 = 1; m2 < 16; m2 <<= 1) mx = fmaxf(mx, __shfl_xor(mx, m2, 32));
      mx = fmaxf(mx, mrow[r]);
      float al = __expf(mrow[r] - mx);
      float p0 = __expf(x0 - mx);
      float p1 = __expf(x1 - mx);
      float rs = p0 + p1;
      for (int m2 = 1; m2 < 16; m2 <<= 1) rs += __shfl_xor(rs, m2, 32);
      mrow[r] = mx;
      lrow[r] = lrow[r] * al + rs;
      for (int t = 0; t < 4; t++) o[t][r] *= al;
      Plds[wave][(r + hi * 8) * 32 + col]      = f2bf(p0);
      Plds[wave][(r + hi * 8) * 32 + col + 16] = f2bf(p1);
    }
    asm volatile("s_wait_dscnt 0" ::: "memory");   // LDS transpose hand-off (same wave)

    // ---- P (16x32) @ V (32x64): 4 column-tile WMMAs, b128 V fragments ----
    Frag16 fp;
    for (int j = 0; j < 8; j++)
      fp.w[j] = ld_u32(&Plds[wave][col * 32 + kbase(j, hi)]);
    for (int t = 0; t < 4; t++) {
      Frag16 fv;
      // transposed V row for head-dim (t*16+col); keys kc..kc+31 contiguous
      const unsigned short* vrow = Vhead + (size_t)(t * 16 + col) * S_ + kc + hi * 8;
      for (int j = 0; j < 4; j++) fv.w[j]     = ld_u32(vrow + 2 * j);        // k=0..7 (+8*hi)
      for (int j = 0; j < 4; j++) fv.w[4 + j] = ld_u32(vrow + 16 + 2 * j);   // k=16..23 (+8*hi)
      o[t] = __builtin_amdgcn_wmma_f32_16x16x32_bf16(
          false, fp.v, false, fv.v, (short)0, o[t], false, false);
    }
  }

  // ---- normalize and write bf16 attn output [b][s][h*64+d] ----
  for (int t = 0; t < 4; t++)
    for (int r = 0; r < 8; r++) {
      int srow = q0 + r + hi * 8;
      float val = o[t][r] / lrow[r];
      Ob[((size_t)(b * S_ + srow)) * (H_ * HD_) + h * HD_ + t * 16 + col] = f2bf(val);
    }
}

// ---------------------------------------------------------------------------
// Host launcher
// ---------------------------------------------------------------------------
extern "C" void kernel_launch(void* const* d_in, const int* in_sizes, int n_in,
                              void* d_out, int out_size, void* d_ws, size_t ws_size,
                              hipStream_t stream) {
  const float* X  = (const float*)d_in[0];
  const float* Wq = (const float*)d_in[1];
  const float* Wk = (const float*)d_in[2];
  const float* Wv = (const float*)d_in[3];
  const float* Wo = (const float*)d_in[4];
  float* out = (float*)d_out;
  (void)in_sizes; (void)n_in; (void)out_size; (void)ws_size;

  const int M = B_ * S_;                 // 4096 token rows
  char* ws = (char*)d_ws;
  size_t off = 0;
  auto carve = [&](size_t bytes) -> char* {
    char* p = ws + off;
    off += (bytes + 255) & ~(size_t)255;
    return p;
  };
  unsigned short* Xb   = (unsigned short*)carve((size_t)M * D_ * 2);
  unsigned short* Wqb  = (unsigned short*)carve((size_t)D_ * (H_ * HD_) * 2);
  unsigned short* Wkb  = (unsigned short*)carve((size_t)D_ * (KV_ * HD_) * 2);
  unsigned short* Wvb  = (unsigned short*)carve((size_t)D_ * (KV_ * HD_) * 2);
  unsigned short* Wob  = (unsigned short*)carve((size_t)(H_ * HD_) * D_ * 2);
  float*          Yq   = (float*)carve((size_t)M * (H_ * HD_) * 4);
  float*          Yk   = (float*)carve((size_t)M * (KV_ * HD_) * 4);
  float*          Yv   = (float*)carve((size_t)M * (KV_ * HD_) * 4);
  unsigned short* Qbuf = (unsigned short*)carve((size_t)B_ * H_  * S_ * HD_ * 2);
  unsigned short* Kbuf = (unsigned short*)carve((size_t)B_ * KV_ * S_ * HD_ * 2);
  unsigned short* Vbuf = (unsigned short*)carve((size_t)B_ * KV_ * S_ * HD_ * 2);
  unsigned short* Obuf = (unsigned short*)carve((size_t)M * (H_ * HD_) * 2);

  // 1) convert inputs to bf16
  {
    int n;
    n = M * D_;              k_f32_to_bf16<<<(n + 255) / 256, 256, 0, stream>>>(X,  Xb,  n);
    n = D_ * H_ * HD_;       k_f32_to_bf16<<<(n + 255) / 256, 256, 0, stream>>>(Wq, Wqb, n);
    n = D_ * KV_ * HD_;      k_f32_to_bf16<<<(n + 255) / 256, 256, 0, stream>>>(Wk, Wkb, n);
    n = D_ * KV_ * HD_;      k_f32_to_bf16<<<(n + 255) / 256, 256, 0, stream>>>(Wv, Wvb, n);
    n = H_ * HD_ * D_;       k_f32_to_bf16<<<(n + 255) / 256, 256, 0, stream>>>(Wo, Wob, n);
  }

  // 2) QKV projections (bf16 WMMA, fp32 results)
  k_gemm_bf16<<<dim3((H_ * HD_) / 64, M / 128), 256, 0, stream>>>(Xb, Wqb, Yq, M, H_ * HD_, D_);
  k_gemm_bf16<<<dim3((KV_ * HD_) / 64, M / 128), 256, 0, stream>>>(Xb, Wkb, Yk, M, KV_ * HD_, D_);
  k_gemm_bf16<<<dim3((KV_ * HD_) / 64, M / 128), 256, 0, stream>>>(Xb, Wvb, Yv, M, KV_ * HD_, D_);

  // 3) RoPE + layout (K head-major, V transposed) to bf16
  {
    int nq = B_ * S_ * H_ * HD_;
    k_rope_q<<<nq / 256, 256, 0, stream>>>(Yq, Qbuf);
    int nk = B_ * S_ * KV_ * HD_;
    k_rope_kv<<<nk / 256, 256, 0, stream>>>(Yk, Yv, Kbuf, Vbuf);
  }

  // 4) windowed flash attention (4 q-tiles per block)
  k_attn<<<(B_ * H_ * (S_ / 16)) / 4, 128, 0, stream>>>(Qbuf, Kbuf, Vbuf, Obuf);

  // 5) output projection -> fp32 d_out
  k_gemm_bf16<<<dim3(D_ / 64, M / 128), 256, 0, stream>>>(Obuf, Wob, out, M, D_, H_ * HD_);
}
